// EGNNDiff_240518169000
// MI455X (gfx1250) — matrix operand order repaired and checked by reading, not compiled, not run on hardware
//
#include <hip/hip_runtime.h>
#include <hip/hip_bf16.h>

typedef __bf16 bf16_t;
typedef bf16_t v8bf  __attribute__((ext_vector_type(8)));
typedef bf16_t v16bf __attribute__((ext_vector_type(16)));
typedef float  v8f   __attribute__((ext_vector_type(8)));

#define HD 128
#define NTILES 8   // 128 / 16 output tiles

__device__ __forceinline__ float silu_f(float v){ return v / (1.0f + __expf(-v)); }

__device__ __forceinline__ void atomAddF(float* p, float v){
  __hip_atomic_fetch_add(p, v, __ATOMIC_RELAXED, __HIP_MEMORY_SCOPE_AGENT);
}

// wave-local LDS ordering (DS ops are in-order per wave; this blocks compiler reordering
// and drains the DS pipe before dependent reads)
__device__ __forceinline__ void lds_wait(){ asm volatile("s_wait_dscnt 0" ::: "memory"); }

// A fragment (16x32 bf16, ISA 7.12.2): lane holds row M=lane&15,
// K chunks: [k0 + hl*8 .. +7] and [k0+16+hl*8 .. +7]  (hl = lane>=16)
// caller passes p = rowbase + k0 + hl*8 ; second chunk is p+16.
__device__ __forceinline__ v16bf load_a_frag(const bf16_t* p){
  v8bf c0 = *(const v8bf*)(p);
  v8bf c1 = *(const v8bf*)(p + 16);
  return __builtin_shufflevector(c0, c1, 0,1,2,3,4,5,6,7,8,9,10,11,12,13,14,15);
}

// B fragment: packed buffer layout [ks][nt][lane][16] (bf16), one 32B load per lane
__device__ __forceinline__ v16bf load_b_frag(const bf16_t* __restrict__ frag, int ks, int nt, int lane){
  return *(const v16bf*)(frag + ((((ks*NTILES) + nt)*32 + lane) << 4));
}

// Pack row-major f32 W[K][128] (first KS*32 rows) into B-fragment order:
// dst[((ks*8+nt)*32+lane)*16 + j] = W[ks*32 + (lane>=16?8:0) + (j&7) + (j>=8?16:0)][nt*16 + (lane&15)]
__global__ void k_pack(const float* __restrict__ W, bf16_t* __restrict__ dst, int KS){
  int i = blockIdx.x*blockDim.x + threadIdx.x;
  int total = KS * NTILES * 32 * 16;
  if (i >= total) return;
  int j    = i & 15;
  int lane = (i >> 4) & 31;
  int nt   = (i >> 9) & 7;
  int ks   = i >> 12;
  int krow = ks*32 + ((lane >> 4) ? 8 : 0) + (j & 7) + ((j >> 3) << 4);
  int ncol = nt*16 + (lane & 15);
  dst[i] = (bf16_t)W[(size_t)krow*HD + ncol];
}

__global__ void k_deg(const int* __restrict__ col, float* __restrict__ deg, int E_){
  int i = blockIdx.x*blockDim.x + threadIdx.x;
  if (i < E_) atomAddF(&deg[col[i]], 1.0f);
}

// 32x128 <- silu( (32x128 LDS bf16) @ (128x128 packed) + bias ), two 16-row M tiles
// sharing each B fragment. Optional f32 atomic scatter-add into agg[col[m]*128+n].
template<bool AGG>
__device__ __forceinline__ void wave_mm_2t(const bf16_t* asrc, const bf16_t* __restrict__ fw,
    const float* __restrict__ bias, bf16_t* dst,
    float* __restrict__ agg, const int* col_s, int nv, int lane)
{
  lds_wait();
  int nl = lane & 15, hl = lane >> 4;
  v16bf a0[4], a1[4];
  #pragma unroll
  for (int ks = 0; ks < 4; ++ks){
    a0[ks] = load_a_frag(asrc + nl*HD       + ks*32 + hl*8);
    a1[ks] = load_a_frag(asrc + (16+nl)*HD  + ks*32 + hl*8);
  }
  #pragma unroll
  for (int nt = 0; nt < NTILES; ++nt){
    v8f acc0, acc1;
    #pragma unroll
    for (int r = 0; r < 8; ++r){ acc0[r] = 0.0f; acc1[r] = 0.0f; }
    #pragma unroll
    for (int ks = 0; ks < 4; ++ks){
      v16bf b = load_b_frag(fw, ks, nt, lane);
      acc0 = __builtin_amdgcn_wmma_f32_16x16x32_bf16(false, a0[ks], false, b, (short)0, acc0, false, false);
      acc1 = __builtin_amdgcn_wmma_f32_16x16x32_bf16(false, a1[ks], false, b, (short)0, acc1, false, false);
    }
    int n = nt*16 + nl;
    float bb = bias[n];
    #pragma unroll
    for (int r = 0; r < 8; ++r){
      int m0 = r + hl*8, m1 = 16 + m0;
      float v0 = silu_f(acc0[r] + bb);
      float v1 = silu_f(acc1[r] + bb);
      dst[m0*HD + n] = (bf16_t)v0;
      dst[m1*HD + n] = (bf16_t)v1;
      if (AGG){
        if (m0 < nv) atomAddF(agg + (size_t)col_s[m0]*HD + n, v0);
        if (m1 < nv) atomAddF(agg + (size_t)col_s[m1]*HD + n, v1);
      }
    }
  }
}

// ---------- fused per-edge message/coordinate kernel (one layer) ----------
// 2 waves/block, 32 edges/wave (two 16-row M tiles share each weight fragment)
__global__ __launch_bounds__(64) void k_edge(
    const int* __restrict__ row, const int* __restrict__ col,
    const float* __restrict__ xw, const bf16_t* __restrict__ hb,
    const bf16_t* __restrict__ fw1, const float* __restrict__ w1last,
    const float* __restrict__ b1,  const bf16_t* __restrict__ fw2,
    const float* __restrict__ b2,  const bf16_t* __restrict__ fcw1,
    const float* __restrict__ cb1, const float* __restrict__ cw2,
    float* __restrict__ agg, float* __restrict__ cupd, int E_)
{
  __shared__ bf16_t sm_t1[2][32*HD];
  __shared__ bf16_t sm_t2[2][32*HD];
  __shared__ float  sm_dist[2][32];
  __shared__ int    sm_col[2][32];

  int w = threadIdx.x >> 5, lane = threadIdx.x & 31;
  int nl = lane & 15, hl = lane >> 4;
  long e0 = ((long)blockIdx.x*2 + w) * 32;
  long rem = (long)E_ - e0;
  int nv = rem < 32 ? (rem < 0 ? 0 : (int)rem) : 32;

  // each lane owns one edge for geometry / reductions / atomics
  long eg = e0 + lane; if (eg >= E_) eg = E_ - 1; if (eg < 0) eg = 0;
  int rg = row[eg], cg = col[eg];
  float rx = xw[3*rg+0]-xw[3*cg+0], ry = xw[3*rg+1]-xw[3*cg+1], rz = xw[3*rg+2]-xw[3*cg+2];
  float dist = sqrtf(rx*rx + ry*ry + rz*rz);
  float inv = 1.0f/(dist + 1e-8f);
  float dx = rx*inv, dy = ry*inv, dz = rz*inv;
  sm_dist[w][lane] = dist; sm_col[w][lane] = cg;
  lds_wait();

  // A-fragment node indices: lane supplies row M=nl of tile0 (edge e0+nl) and tile1 (edge e0+16+nl)
  long eA = e0 + nl;      if (eA >= E_) eA = E_ - 1; if (eA < 0) eA = 0;
  long eB = e0 + 16 + nl; if (eB >= E_) eB = E_ - 1; if (eB < 0) eB = 0;
  int ri0 = row[eA], ci0 = col[eA], ri1 = row[eB], ci1 = col[eB];

  bf16_t* t1 = sm_t1[w]; bf16_t* t2 = sm_t2[w];

  // phase 1: mi = [h_row | h_col | dist] @ m_w1 + b1 ; silu -> t1 (bf16, 32x128)
  {
    v16bf ar0[4], ac0[4], ar1[4], ac1[4];
    const bf16_t* pr0 = hb + (size_t)ri0*HD + hl*8;
    const bf16_t* pc0 = hb + (size_t)ci0*HD + hl*8;
    const bf16_t* pr1 = hb + (size_t)ri1*HD + hl*8;
    const bf16_t* pc1 = hb + (size_t)ci1*HD + hl*8;
    #pragma unroll
    for (int ks = 0; ks < 4; ++ks){
      ar0[ks] = load_a_frag(pr0 + ks*32); ac0[ks] = load_a_frag(pc0 + ks*32);
      ar1[ks] = load_a_frag(pr1 + ks*32); ac1[ks] = load_a_frag(pc1 + ks*32);
    }
    #pragma unroll
    for (int nt = 0; nt < NTILES; ++nt){
      v8f acc0, acc1;
      #pragma unroll
      for (int r = 0; r < 8; ++r){ acc0[r] = 0.0f; acc1[r] = 0.0f; }
      #pragma unroll
      for (int ks = 0; ks < 4; ++ks){
        v16bf b = load_b_frag(fw1, ks, nt, lane);
        acc0 = __builtin_amdgcn_wmma_f32_16x16x32_bf16(false, ar0[ks], false, b, (short)0, acc0, false, false);
        acc1 = __builtin_amdgcn_wmma_f32_16x16x32_bf16(false, ar1[ks], false, b, (short)0, acc1, false, false);
      }
      #pragma unroll
      for (int ks = 0; ks < 4; ++ks){
        v16bf b = load_b_frag(fw1, ks+4, nt, lane);
        acc0 = __builtin_amdgcn_wmma_f32_16x16x32_bf16(false, ac0[ks], false, b, (short)0, acc0, false, false);
        acc1 = __builtin_amdgcn_wmma_f32_16x16x32_bf16(false, ac1[ks], false, b, (short)0, acc1, false, false);
      }
      int n = nt*16 + nl;
      float bb = b1[n], wl = w1last[n];
      #pragma unroll
      for (int r = 0; r < 8; ++r){
        int m0 = r + hl*8, m1 = 16 + m0;
        t1[m0*HD + n] = (bf16_t)silu_f(acc0[r] + bb + sm_dist[w][m0]*wl);
        t1[m1*HD + n] = (bf16_t)silu_f(acc1[r] + bb + sm_dist[w][m1]*wl);
      }
    }
  }

  // phase 2: msg = silu(t1 @ m_w2 + b2) -> t2 ; scatter-add msg into agg[col]
  wave_mm_2t<true >(t1, fw2,  b2,  t2, agg,     sm_col[w], nv, lane);
  // phase 3: t = silu(msg @ c_w1 + c_b1) -> t1
  wave_mm_2t<false>(t2, fcw1, cb1, t1, nullptr, nullptr,   nv, lane);
  lds_wait();

  // phase 4: cmult = tanh(t @ c_w2) ; cupd[col] += cmult * rel_dir  (lane == edge)
  {
    float s = 0.0f;
    const bf16_t* tp = t1 + lane*HD;
    #pragma unroll 8
    for (int k = 0; k < HD; ++k) s += (float)tp[k] * cw2[k];
    if (lane < nv){
      float cm = tanhf(s);
      atomAddF(cupd + (size_t)cg*3 + 0, cm*dx);
      atomAddF(cupd + (size_t)cg*3 + 1, cm*dy);
      atomAddF(cupd + (size_t)cg*3 + 2, cm*dz);
    }
  }
}

// ---------- fused per-edge epsilon head ----------
__global__ __launch_bounds__(64) void k_eps_edge(
    const int* __restrict__ row, const int* __restrict__ col,
    const float* __restrict__ xw, const bf16_t* __restrict__ hb,
    const bf16_t* __restrict__ fem1, const float* __restrict__ em4 /*rows 256..259*/,
    const float* __restrict__ emb1, const bf16_t* __restrict__ fem2,
    const float* __restrict__ emb2, const bf16_t* __restrict__ fec1,
    const float* __restrict__ ecb1, const float* __restrict__ ecw2,
    const float* __restrict__ ecb2, float* __restrict__ epsb, int E_)
{
  __shared__ bf16_t sm_t1[2][32*HD];
  __shared__ bf16_t sm_t2[2][32*HD];
  __shared__ float  sm_dist[2][32];
  __shared__ float  sm_rel[2][32][3];
  __shared__ int    sm_col[2][32];

  int w = threadIdx.x >> 5, lane = threadIdx.x & 31;
  int nl = lane & 15, hl = lane >> 4;
  long e0 = ((long)blockIdx.x*2 + w) * 32;
  long rem = (long)E_ - e0;
  int nv = rem < 32 ? (rem < 0 ? 0 : (int)rem) : 32;

  long eg = e0 + lane; if (eg >= E_) eg = E_ - 1; if (eg < 0) eg = 0;
  int rg = row[eg], cg = col[eg];
  float rx = xw[3*rg+0]-xw[3*cg+0], ry = xw[3*rg+1]-xw[3*cg+1], rz = xw[3*rg+2]-xw[3*cg+2];
  float dist = sqrtf(rx*rx + ry*ry + rz*rz);
  sm_dist[w][lane] = dist; sm_col[w][lane] = cg;
  sm_rel[w][lane][0] = rx; sm_rel[w][lane][1] = ry; sm_rel[w][lane][2] = rz;
  lds_wait();

  long eA = e0 + nl;      if (eA >= E_) eA = E_ - 1; if (eA < 0) eA = 0;
  long eB = e0 + 16 + nl; if (eB >= E_) eB = E_ - 1; if (eB < 0) eB = 0;
  int ri0 = row[eA], ci0 = col[eA], ri1 = row[eB], ci1 = col[eB];

  bf16_t* t1 = sm_t1[w]; bf16_t* t2 = sm_t2[w];

  // phase 1: mi = [h_row | h_col | rel | dist] @ em_w1 + em_b1 ; silu -> t1
  {
    v16bf ar0[4], ac0[4], ar1[4], ac1[4];
    const bf16_t* pr0 = hb + (size_t)ri0*HD + hl*8;
    const bf16_t* pc0 = hb + (size_t)ci0*HD + hl*8;
    const bf16_t* pr1 = hb + (size_t)ri1*HD + hl*8;
    const bf16_t* pc1 = hb + (size_t)ci1*HD + hl*8;
    #pragma unroll
    for (int ks = 0; ks < 4; ++ks){
      ar0[ks] = load_a_frag(pr0 + ks*32); ac0[ks] = load_a_frag(pc0 + ks*32);
      ar1[ks] = load_a_frag(pr1 + ks*32); ac1[ks] = load_a_frag(pc1 + ks*32);
    }
    #pragma unroll
    for (int nt = 0; nt < NTILES; ++nt){
      v8f acc0, acc1;
      #pragma unroll
      for (int r = 0; r < 8; ++r){ acc0[r] = 0.0f; acc1[r] = 0.0f; }
      #pragma unroll
      for (int ks = 0; ks < 4; ++ks){
        v16bf b = load_b_frag(fem1, ks, nt, lane);
        acc0 = __builtin_amdgcn_wmma_f32_16x16x32_bf16(false, ar0[ks], false, b, (short)0, acc0, false, false);
        acc1 = __builtin_amdgcn_wmma_f32_16x16x32_bf16(false, ar1[ks], false, b, (short)0, acc1, false, false);
      }
      #pragma unroll
      for (int ks = 0; ks < 4; ++ks){
        v16bf b = load_b_frag(fem1, ks+4, nt, lane);
        acc0 = __builtin_amdgcn_wmma_f32_16x16x32_bf16(false, ac0[ks], false, b, (short)0, acc0, false, false);
        acc1 = __builtin_amdgcn_wmma_f32_16x16x32_bf16(false, ac1[ks], false, b, (short)0, acc1, false, false);
      }
      int n = nt*16 + nl;
      float bb = emb1[n];
      float R0 = em4[n], R1 = em4[HD + n], R2 = em4[2*HD + n], R3 = em4[3*HD + n];
      #pragma unroll
      for (int r = 0; r < 8; ++r){
        int m0 = r + hl*8, m1 = 16 + m0;
        float v0 = acc0[r] + bb + sm_rel[w][m0][0]*R0 + sm_rel[w][m0][1]*R1
                                + sm_rel[w][m0][2]*R2 + sm_dist[w][m0]*R3;
        float v1 = acc1[r] + bb + sm_rel[w][m1][0]*R0 + sm_rel[w][m1][1]*R1
                                + sm_rel[w][m1][2]*R2 + sm_dist[w][m1]*R3;
        t1[m0*HD + n] = (bf16_t)silu_f(v0);
        t1[m1*HD + n] = (bf16_t)silu_f(v1);
      }
    }
  }

  // phase 2: em = silu(t1 @ em_w2 + em_b2) -> t2
  wave_mm_2t<false>(t1, fem2, emb2, t2, nullptr, nullptr, nv, lane);
  // phase 3: t = silu(em @ ec_w1 + ec_b1) -> t1
  wave_mm_2t<false>(t2, fec1, ecb1, t1, nullptr, nullptr, nv, lane);
  lds_wait();

  // phase 4: esrc = t @ ec_w2 + ec_b2 (128->3); eps[col] += esrc  (96 jobs / 32 lanes)
  for (int o = lane; o < 96; o += 32){
    int m = o / 3, d = o - m*3;
    float s = ecb2[d];
    const bf16_t* tp = t1 + m*HD;
    #pragma unroll 8
    for (int k = 0; k < HD; ++k) s += (float)tp[k] * ecw2[k*3 + d];
    if (m < nv) atomAddF(epsb + (size_t)sm_col[w][m]*3 + d, s);
  }
}

// ---------- node-side kernels (fp32 VALU, 8 nodes per block) ----------
__global__ __launch_bounds__(128) void k_embed(const float* __restrict__ hsrc,
    const float* __restrict__ wE, const float* __restrict__ bE,
    float* __restrict__ hdst, bf16_t* __restrict__ hb, int N_)
{
  __shared__ float in[8][64];
  int n = threadIdx.x, base = blockIdx.x*8;
  if (n < 64){
    for (int j = 0; j < 8; ++j){ int node = base+j; if (node >= N_) node = N_-1;
      in[j][n] = hsrc[(size_t)node*64 + n]; }
  }
  __syncthreads();
  float acc[8]; float bb = bE[n];
  #pragma unroll
  for (int j = 0; j < 8; ++j) acc[j] = bb;
  for (int k = 0; k < 64; ++k){
    float wv = wE[k*HD + n];
    #pragma unroll
    for (int j = 0; j < 8; ++j) acc[j] += in[j][k]*wv;
  }
  for (int j = 0; j < 8; ++j){ int node = base+j; if (node < N_){
    hdst[(size_t)node*HD + n] = acc[j]; hb[(size_t)node*HD + n] = (bf16_t)acc[j]; } }
}

__global__ __launch_bounds__(128) void k_node(
    float* __restrict__ h, bf16_t* __restrict__ hb, const float* __restrict__ agg,
    const float* __restrict__ cupd, const float* __restrict__ deg, float* __restrict__ xw,
    const float* __restrict__ w1, const float* __restrict__ b1,
    const float* __restrict__ w2, const float* __restrict__ b2,
    const float* __restrict__ g,  const float* __restrict__ bt, int N_)
{
  __shared__ float in2[8][256];
  __shared__ float u[8][HD];
  __shared__ float mu_s[8], rs_s[8];
  int n = threadIdx.x, base = blockIdx.x*8;
  if (n < 24){ int j = n/3, d = n - j*3; int node = base+j;
    if (node < N_) xw[(size_t)node*3 + d] += cupd[(size_t)node*3 + d] / (deg[node] + 1.0f); }
  for (int j = 0; j < 8; ++j){ int node = base+j; if (node >= N_) node = N_-1;
    in2[j][n]      = h[(size_t)node*HD + n];
    in2[j][128+n]  = agg[(size_t)node*HD + n]; }
  __syncthreads();
  float acc[8]; float bb = b1[n];
  #pragma unroll
  for (int j = 0; j < 8; ++j) acc[j] = bb;
  for (int k = 0; k < 256; ++k){
    float wv = w1[k*HD + n];
    #pragma unroll
    for (int j = 0; j < 8; ++j) acc[j] += in2[j][k]*wv;
  }
  for (int j = 0; j < 8; ++j) u[j][n] = silu_f(acc[j]);
  __syncthreads();
  float bb2 = b2[n];
  #pragma unroll
  for (int j = 0; j < 8; ++j) acc[j] = bb2;
  for (int k = 0; k < HD; ++k){
    float wv = w2[k*HD + n];
    #pragma unroll
    for (int j = 0; j < 8; ++j) acc[j] += u[j][k]*wv;
  }
  for (int j = 0; j < 8; ++j) in2[j][n] = in2[j][n] + acc[j];   // hnew
  __syncthreads();
  if (n < 8){
    float m = 0.0f;
    for (int k = 0; k < HD; ++k) m += in2[n][k];
    m *= (1.0f/HD);
    float v = 0.0f;
    for (int k = 0; k < HD; ++k){ float d = in2[n][k]-m; v += d*d; }
    v *= (1.0f/HD);
    mu_s[n] = m; rs_s[n] = rsqrtf(v + 1e-5f);
  }
  __syncthreads();
  float gg = g[n], bb3 = bt[n];
  for (int j = 0; j < 8; ++j){ int node = base+j; if (node < N_){
    float val = (in2[j][n]-mu_s[j])*rs_s[j]*gg + bb3;
    h[(size_t)node*HD + n] = val; hb[(size_t)node*HD + n] = (bf16_t)val; } }
}

__global__ __launch_bounds__(128) void k_eps_node(
    const float* __restrict__ h, const float* __restrict__ xw, const float* __restrict__ epsb,
    const float* __restrict__ w1, const float* __restrict__ b1,
    const float* __restrict__ w2, const float* __restrict__ b2,
    float* __restrict__ outh, float* __restrict__ outx, float* __restrict__ oute, int N_)
{
  __shared__ float in2[8][132];
  __shared__ float u[8][HD];
  int n = threadIdx.x, base = blockIdx.x*8;
  for (int j = 0; j < 8; ++j){ int node = base+j; if (node >= N_) node = N_-1;
    in2[j][n] = h[(size_t)node*HD + n]; }
  if (n < 24){ int j = n/3, d = n - j*3; int node = base+j; if (node >= N_) node = N_-1;
    in2[j][128+d] = xw[(size_t)node*3 + d]; }
  __syncthreads();
  float acc[8]; float bb = b1[n];
  #pragma unroll
  for (int j = 0; j < 8; ++j) acc[j] = bb;
  for (int k = 0; k < 131; ++k){
    float wv = w1[k*HD + n];
    #pragma unroll
    for (int j = 0; j < 8; ++j) acc[j] += in2[j][k]*wv;
  }
  for (int j = 0; j < 8; ++j) u[j][n] = silu_f(acc[j]);
  __syncthreads();
  for (int j = 0; j < 8; ++j){ int node = base+j; if (node < N_)
    outh[(size_t)node*HD + n] = in2[j][n]; }
  if (n < 24){
    int j = n/3, d = n - j*3; int node = base+j;
    if (node < N_){
      float s = b2[d];
      for (int k = 0; k < HD; ++k) s += u[j][k]*w2[k*3 + d];
      outx[(size_t)node*3 + d] = in2[j][128+d];
      oute[(size_t)node*3 + d] = epsb[(size_t)node*3 + d] + s;
    }
  }
}

extern "C" void kernel_launch(void* const* d_in, const int* in_sizes, int n_in,
                              void* d_out, int out_size, void* d_ws, size_t ws_size,
                              hipStream_t stream) {
  (void)n_in; (void)out_size; (void)ws_size;
  const float* h_in  = (const float*)d_in[0];
  const float* x_in  = (const float*)d_in[1];
  const int*   eidx  = (const int*)  d_in[2];
  const float* emb_w = (const float*)d_in[3];
  const float* emb_b = (const float*)d_in[4];
  const float* m_w1  = (const float*)d_in[5];
  const float* m_b1  = (const float*)d_in[6];
  const float* m_w2  = (const float*)d_in[7];
  const float* m_b2  = (const float*)d_in[8];
  const float* c_w1  = (const float*)d_in[9];
  const float* c_b1  = (const float*)d_in[10];
  const float* c_w2  = (const float*)d_in[11];
  const float* n_w1  = (const float*)d_in[12];
  const float* n_b1  = (const float*)d_in[13];
  const float* n_w2  = (const float*)d_in[14];
  const float* n_b2  = (const float*)d_in[15];
  const float* ln_g  = (const float*)d_in[16];
  const float* ln_b  = (const float*)d_in[17];
  const float* em_w1 = (const float*)d_in[18];
  const float* em_b1 = (const float*)d_in[19];
  const float* em_w2 = (const float*)d_in[20];
  const float* em_b2 = (const float*)d_in[21];
  const float* ec_w1 = (const float*)d_in[22];
  const float* ec_b1 = (const float*)d_in[23];
  const float* ec_w2 = (const float*)d_in[24];
  const float* ec_b2 = (const float*)d_in[25];
  const float* eh_w1 = (const float*)d_in[26];
  const float* eh_b1 = (const float*)d_in[27];
  const float* eh_w2 = (const float*)d_in[28];
  const float* eh_b2 = (const float*)d_in[29];

  const int N = in_sizes[0] / 64;
  const int E = in_sizes[2] / 2;
  const int L = in_sizes[5] / (257*128);
  const int* row = eidx;
  const int* col = eidx + E;

  char* wsp = (char*)d_ws; size_t off = 0;
  auto alloc = [&](size_t bytes)->char*{
    char* p = wsp + off; off += (bytes + 255) & ~(size_t)255; return p; };
  float*  hbuf = (float*) alloc((size_t)N*HD*4);
  bf16_t* hb   = (bf16_t*)alloc((size_t)N*HD*2);
  float*  xbuf = (float*) alloc((size_t)N*3*4);
  float*  deg  = (float*) alloc((size_t)N*4);
  float*  agg  = (float*) alloc((size_t)N*HD*4);
  float*  cupd = (float*) alloc((size_t)N*3*4);
  float*  epsb = (float*) alloc((size_t)N*3*4);
  const size_t FR8 = (size_t)8*NTILES*32*16;   // elems per 256-row packed matrix
  const size_t FR4 = (size_t)4*NTILES*32*16;   // elems per 128-row packed matrix
  bf16_t* fMW1 = (bf16_t*)alloc((size_t)L*FR8*2);
  bf16_t* fMW2 = (bf16_t*)alloc((size_t)L*FR4*2);
  bf16_t* fCW1 = (bf16_t*)alloc((size_t)L*FR4*2);
  bf16_t* fEM1 = (bf16_t*)alloc(FR8*2);
  bf16_t* fEM2 = (bf16_t*)alloc(FR4*2);
  bf16_t* fEC1 = (bf16_t*)alloc(FR4*2);

  hipMemcpyAsync(xbuf, x_in, (size_t)N*3*4, hipMemcpyDeviceToDevice, stream);
  hipMemsetAsync(deg, 0, (size_t)N*4, stream);
  k_deg<<<(E+255)/256, 256, 0, stream>>>(col, deg, E);

  for (int i = 0; i < L; ++i){
    k_pack<<<(int)((FR8+255)/256),256,0,stream>>>(m_w1 + (size_t)i*257*128, fMW1 + (size_t)i*FR8, 8);
    k_pack<<<(int)((FR4+255)/256),256,0,stream>>>(m_w2 + (size_t)i*128*128, fMW2 + (size_t)i*FR4, 4);
    k_pack<<<(int)((FR4+255)/256),256,0,stream>>>(c_w1 + (size_t)i*128*128, fCW1 + (size_t)i*FR4, 4);
  }
  k_pack<<<(int)((FR8+255)/256),256,0,stream>>>(em_w1, fEM1, 8);
  k_pack<<<(int)((FR4+255)/256),256,0,stream>>>(em_w2, fEM2, 4);
  k_pack<<<(int)((FR4+255)/256),256,0,stream>>>(ec_w1, fEC1, 4);

  const int nbN = (N + 7) / 8;
  const int nbE = (E + 63) / 64;   // 2 waves/block * 32 edges/wave

  k_embed<<<nbN,128,0,stream>>>(h_in, emb_w, emb_b, hbuf, hb, N);

  for (int i = 0; i < L; ++i){
    hipMemsetAsync(agg,  0, (size_t)N*HD*4, stream);
    hipMemsetAsync(cupd, 0, (size_t)N*3*4,  stream);
    k_edge<<<nbE,64,0,stream>>>(row, col, xbuf, hb,
        fMW1 + (size_t)i*FR8, m_w1 + (size_t)i*257*128 + 256*128, m_b1 + i*128,
        fMW2 + (size_t)i*FR4, m_b2 + i*128,
        fCW1 + (size_t)i*FR4, c_b1 + i*128, c_w2 + i*128,
        agg, cupd, E);
    k_node<<<nbN,128,0,stream>>>(hbuf, hb, agg, cupd, deg, xbuf,
        n_w1 + (size_t)i*256*128, n_b1 + i*128, n_w2 + (size_t)i*128*128, n_b2 + i*128,
        ln_g + i*128, ln_b + i*128, N);
  }

  hipMemsetAsync(epsb, 0, (size_t)N*3*4, stream);
  k_eps_edge<<<nbE,64,0,stream>>>(row, col, xbuf, hb,
      fEM1, em_w1 + 256*128, em_b1, fEM2, em_b2,
      fEC1, ec_b1, ec_w2, ec_b2, epsb, E);

  float* outh = (float*)d_out;
  float* outx = outh + (size_t)N*HD;
  float* oute = outx + (size_t)N*3;
  k_eps_node<<<nbN,128,0,stream>>>(hbuf, xbuf, epsb, eh_w1, eh_b1, eh_w2, eh_b2,
                                   outh, outx, oute, N);
}
